// CustomLSTM_22368189678462
// MI455X (gfx1250) — compile-verified, matrix operand order checked by reading, not run on hardware
//
#include <hip/hip_runtime.h>

#define B 64
#define S 1024
#define I0 256
#define H 512
#define G 2048      // 4*H gate columns
#define CHUNK 128   // timesteps per chunk
#define NWG 16      // persistent workgroups
#define HC 32       // h-columns owned per workgroup
#define KP 40       // LDS k-stride (pad) for GEMM tiles, 80B rows -> conflict-free & 16B aligned
#define WKP 520     // LDS k-stride for persistent kernel (512 + 8)

typedef __attribute__((ext_vector_type(16))) __bf16 v16bf;
typedef __attribute__((ext_vector_type(8)))  float  v8f;

// -------- CDNA5 async global->LDS copy (ASYNCcnt-tracked) -------------------
// Copies 16 bytes from a global address straight into LDS, no VGPR round-trip.
__device__ inline void async_g2l_b128(const void* lds_ptr, const void* gptr) {
    unsigned ldsoff = (unsigned)(unsigned long long)lds_ptr;  // LDS aperture: addr[31:0]
    asm volatile("global_load_async_to_lds_b128 %0, %1, off"
                 :: "v"(ldsoff), "v"(gptr)
                 : "memory");
}
__device__ inline void wait_async0() {
    asm volatile("s_wait_asynccnt 0x0" ::: "memory");
}

// -------- WMMA fragment gather from LDS ------------------------------------
// A (16x32) / B (32x16) bf16 fragment, 16 elems per lane.
// element j: K = (j/8)*16 + hi*8 + (j%8)  ; rc = row (A) or col (B) = lane&15
__device__ inline v16bf frag_ld(const __bf16* base, int stride, int rc, int hi) {
    const __bf16* p = base + rc * stride + hi * 8;
    v16bf v;
#pragma unroll
    for (int j = 0; j < 8; ++j) {
        v[j]     = p[j];
        v[j + 8] = p[16 + j];
    }
    return v;
}

__device__ inline v8f v8f_zero() {
    v8f z;
#pragma unroll
    for (int j = 0; j < 8; ++j) z[j] = 0.0f;
    return z;
}

__device__ inline float sigm(float x) { return 1.0f / (1.0f + __expf(-x)); }

// -------- packing kernels ---------------------------------------------------
__global__ void k_f32_to_bf16(const float* __restrict__ src, __bf16* __restrict__ dst, int n) {
    for (int i = blockIdx.x * blockDim.x + threadIdx.x; i < n; i += gridDim.x * blockDim.x)
        dst[i] = (__bf16)src[i];
}

// Wsrc: [G][Kdim] row-major  ->  Wt: [Kdim][G] bf16 (K-major)
__global__ void k_pack_wT(const float* __restrict__ Wsrc, __bf16* __restrict__ Wt, int Kdim) {
    int n = Kdim * G;
    for (int i = blockIdx.x * blockDim.x + threadIdx.x; i < n; i += gridDim.x * blockDim.x) {
        int k = i / G, g = i % G;
        Wt[i] = (__bf16)Wsrc[g * Kdim + k];
    }
}

__global__ void k_bias(const float* __restrict__ bi, const float* __restrict__ bh,
                       float* __restrict__ bias) {
    int i = blockIdx.x * blockDim.x + threadIdx.x;
    if (i < G) bias[i] = bi[i] + bh[i];
}

// -------- chunk input GEMM:  xg[sl*B+b][n] = x[b, s0+sl, :] . Wt[:, n] + bias[n]
// M = CHUNK*B = 8192 rows (time-major), N = G = 2048, K = Kdim (256 or 512)
#define MT 64
#define NT 256
__global__ __launch_bounds__(256) void k_gemm_xg(const __bf16* __restrict__ X,
                                                 const __bf16* __restrict__ Wt,
                                                 const float* __restrict__ bias,
                                                 float* __restrict__ xg,
                                                 int Kdim, int s0) {
    __shared__ __bf16 As[MT * KP];   // [row][k]
    __shared__ __bf16 Bs[NT * KP];   // [col][k]
    const int tid = threadIdx.x;
    const int lane = tid & 31, wave = tid >> 5;
    const int hi = lane >> 4, lr = lane & 15;
    const int m0 = blockIdx.x * MT;
    const int n0 = blockIdx.y * NT;
    const int wm = wave & 3;            // M tile 0..3
    const int wn = (wave >> 2) * 128;   // N half offset

    v8f acc[8];
#pragma unroll
    for (int t = 0; t < 8; ++t) acc[t] = v8f_zero();

    for (int kt = 0; kt < Kdim; kt += 32) {
        // A tile: 64 rows x 32 k, one async b128 (8 bf16) per thread, straight to LDS
        {
            int row = tid >> 2;
            int kc  = (tid & 3) * 8;
            int r   = m0 + row;
            int b   = r & (B - 1);
            int sl  = r >> 6;
            const __bf16* src = X + (size_t)(b * S + s0 + sl) * Kdim + kt + kc;
            if (kt + 32 < Kdim) __builtin_prefetch(src + 32, 0, 0);
            async_g2l_b128(&As[row * KP + kc], src);
        }
        // B tile: 256 cols x 32 k ; coalesced over n, register-transposed into [col][k]
#pragma unroll
        for (int it = 0; it < 4; ++it) {
            int cid = tid + it * 256;
            int k   = cid >> 5;          // 0..31
            int nc  = (cid & 31) * 8;    // 0..248
            const __bf16* src = Wt + (size_t)(kt + k) * G + n0 + nc;
            uint4 d = *(const uint4*)src;
            const __bf16* e = (const __bf16*)&d;
#pragma unroll
            for (int j = 0; j < 8; ++j) Bs[(nc + j) * KP + k] = e[j];
        }
        wait_async0();
        __syncthreads();

        v16bf af = frag_ld(&As[(wm * 16) * KP], KP, lr, hi);
#pragma unroll
        for (int t = 0; t < 8; ++t) {
            v16bf bfr = frag_ld(&Bs[(wn + t * 16) * KP], KP, lr, hi);
            acc[t] = __builtin_amdgcn_wmma_f32_16x16x32_bf16(
                false, af, false, bfr, (short)0, acc[t], false, false);
        }
        __syncthreads();
    }

    // epilogue: add bias, store fp32 time-major
#pragma unroll
    for (int t = 0; t < 8; ++t) {
        int colg = n0 + wn + t * 16 + lr;
        float bv = bias[colg];
#pragma unroll
        for (int r = 0; r < 8; ++r) {
            int row = m0 + wm * 16 + hi * 8 + r;
            xg[(size_t)row * G + colg] = acc[t][r] + bv;
        }
    }
}

// -------- persistent recurrent kernel ---------------------------------------
// 16 WGs, each owns HC=32 h-columns -> 128 packed gate columns.
// W_hh slice stays in LDS for all CHUNK steps; h ping-pongs via L2;
// c lives in LDS; one grid barrier per timestep.
__global__ __launch_bounds__(256) void k_lstm_chunk(
    const float* __restrict__ xg,          // [CHUNK*B][G] time-major
    const __bf16* __restrict__ Whh,        // [H][G] K-major
    __bf16* __restrict__ hbuf,             // [2][B*H] ping-pong
    float* __restrict__ cstate,            // [B*H]
    __bf16* __restrict__ outbf,            // layer0: bf16 out -> next layer input
    float* __restrict__ outf,              // layer1: fp32 out -> d_out
    float* __restrict__ dstates,           // d_out tail: h_n[2][B][H], c_n[2][B][H]
    unsigned* __restrict__ counter, int s0, int layer) {
    extern __shared__ char smem[];
    __bf16* Wls = (__bf16*)smem;                                 // [128][WKP]
    __bf16* Hls = (__bf16*)(smem + 128 * WKP * 2);               // [B][WKP]
    float*  Cls = (float*)(smem + 128 * WKP * 2 + B * WKP * 2);  // [B][HC]

    const int tid = threadIdx.x, lane = tid & 31, wave = tid >> 5;
    const int hi = lane >> 4, lr = lane & 15;
    const int wg = blockIdx.x;
    const int hc0 = wg * HC;

    // pack W_hh slice: packed col p = g*32 + cc  ->  global gate col g*H + hc0 + cc
    for (int it = tid; it < 512 * 16; it += 256) {
        int k   = it >> 4;
        int cg  = it & 15;
        int p0  = cg * 8;
        int g   = p0 >> 5;
        int cc0 = p0 & 31;
        uint4 d = *(const uint4*)(Whh + (size_t)k * G + g * H + hc0 + cc0);
        const __bf16* e = (const __bf16*)&d;
#pragma unroll
        for (int j = 0; j < 8; ++j) Wls[(p0 + j) * WKP + k] = e[j];
    }
    // load persistent c slice
    for (int it = tid; it < B * HC; it += 256) {
        int b = it >> 5, cc = it & 31;
        Cls[b * HC + cc] = cstate[b * H + hc0 + cc];
    }
    __syncthreads();

    const int mt = wave >> 1;   // batch tile 0..3 (16 rows each)
    const int ct = wave & 1;    // local col half (16 cols each)
    unsigned gen = 0;

    for (int sl = 0; sl < CHUNK; ++sl) {
        const int s = s0 + sl;
        const __bf16* hsrc = hbuf + (size_t)(s & 1) * (B * H);
        __bf16*       hdst = hbuf + (size_t)((s + 1) & 1) * (B * H);

        // snapshot h (B x H bf16) into LDS via async copies (no VGPR round-trip)
        for (int it = tid; it < (B * H) / 8; it += 256) {
            int b  = it >> 6;
            int kc = (it & 63) * 8;
            async_g2l_b128(&Hls[b * WKP + kc], hsrc + b * H + kc);
        }
        wait_async0();
        __syncthreads();

        v8f acc[4];
#pragma unroll
        for (int g = 0; g < 4; ++g) acc[g] = v8f_zero();

#pragma unroll 4
        for (int kt = 0; kt < H; kt += 32) {
            v16bf af = frag_ld(Hls + (mt * 16) * WKP + kt, WKP, lr, hi);
#pragma unroll
            for (int g = 0; g < 4; ++g) {
                v16bf bfr = frag_ld(Wls + (g * 32 + ct * 16) * WKP + kt, WKP, lr, hi);
                acc[g] = __builtin_amdgcn_wmma_f32_16x16x32_bf16(
                    false, af, false, bfr, (short)0, acc[g], false, false);
            }
        }

        // elementwise LSTM cell: each lane owns (b = mt*16+hi*8+r, col = hc0+ct*16+lr)
        const int cc  = ct * 16 + lr;
        const int col = hc0 + cc;
#pragma unroll
        for (int r = 0; r < 8; ++r) {
            int b = mt * 16 + hi * 8 + r;
            const float* xr = xg + (size_t)(sl * B + b) * G;
            float iv = acc[0][r] + xr[col];
            float fv = acc[1][r] + xr[H + col];
            float gv = acc[2][r] + xr[2 * H + col];
            float ov = acc[3][r] + xr[3 * H + col];
            float ig = sigm(iv);
            float fg = sigm(fv);
            float gg = tanhf(gv);
            float og = sigm(ov);
            float c  = Cls[b * HC + cc];
            float cn = fg * c + ig * gg;
            float hn = og * tanhf(cn);
            Cls[b * HC + cc] = cn;
            hdst[b * H + col] = (__bf16)hn;
            if (outbf) outbf[((size_t)b * S + s) * H + col] = (__bf16)hn;
            if (outf)  outf[((size_t)b * S + s) * H + col] = hn;
            if (s == S - 1) {
                dstates[(size_t)layer * B * H + b * H + col]       = hn;  // h_n
                dstates[(size_t)(2 + layer) * B * H + b * H + col] = cn;  // c_n
            }
        }

        // device-wide barrier (monotonic counter, generation target)
        __syncthreads();
        ++gen;
        if (tid == 0) {
            __threadfence();
            atomicAdd(counter, 1u);
            unsigned target = gen * NWG;
            while (*(volatile unsigned*)counter < target) __builtin_amdgcn_s_sleep(1);
        }
        __syncthreads();
    }

    // persist c slice for next chunk
    for (int it = tid; it < B * HC; it += 256) {
        int b = it >> 5, cc = it & 31;
        cstate[b * H + hc0 + cc] = Cls[b * HC + cc];
    }
}

// -------- host --------------------------------------------------------------
extern "C" void kernel_launch(void* const* d_in, const int* in_sizes, int n_in,
                              void* d_out, int out_size, void* d_ws, size_t ws_size,
                              hipStream_t stream) {
    const float* x    = (const float*)d_in[0];
    const float* Wih0 = (const float*)d_in[1];
    const float* Whh0 = (const float*)d_in[2];
    const float* bih0 = (const float*)d_in[3];
    const float* bhh0 = (const float*)d_in[4];
    const float* Wih1 = (const float*)d_in[5];
    const float* Whh1 = (const float*)d_in[6];
    const float* bih1 = (const float*)d_in[7];
    const float* bhh1 = (const float*)d_in[8];
    float* out = (float*)d_out;

    char* w = (char*)d_ws;
    size_t off = 0;
    auto take = [&](size_t bytes) -> char* {
        char* p = w + off;
        off = (off + bytes + 255) & ~(size_t)255;
        return p;
    };
    __bf16* xbf0   = (__bf16*)take((size_t)B * S * I0 * 2);
    __bf16* xbf1   = (__bf16*)take((size_t)B * S * H * 2);
    __bf16* Wt_ih0 = (__bf16*)take((size_t)I0 * G * 2);
    __bf16* Wt_hh0 = (__bf16*)take((size_t)H * G * 2);
    __bf16* Wt_ih1 = (__bf16*)take((size_t)H * G * 2);
    __bf16* Wt_hh1 = (__bf16*)take((size_t)H * G * 2);
    float*  bias0  = (float*)take((size_t)G * 4);
    float*  bias1  = (float*)take((size_t)G * 4);
    float*  xg     = (float*)take((size_t)CHUNK * B * G * 4);
    __bf16* hbuf   = (__bf16*)take((size_t)2 * B * H * 2);
    float*  cst    = (float*)take((size_t)B * H * 4);
    unsigned* counter = (unsigned*)take(256);

    // one-time packing / conversion
    k_f32_to_bf16<<<512, 256, 0, stream>>>(x, xbf0, B * S * I0);
    k_pack_wT<<<512, 256, 0, stream>>>(Wih0, Wt_ih0, I0);
    k_pack_wT<<<512, 256, 0, stream>>>(Whh0, Wt_hh0, H);
    k_pack_wT<<<512, 256, 0, stream>>>(Wih1, Wt_ih1, H);
    k_pack_wT<<<512, 256, 0, stream>>>(Whh1, Wt_hh1, H);
    k_bias<<<(G + 255) / 256, 256, 0, stream>>>(bih0, bhh0, bias0);
    k_bias<<<(G + 255) / 256, 256, 0, stream>>>(bih1, bhh1, bias1);

    const size_t ldsBytes = (size_t)128 * WKP * 2 + (size_t)B * WKP * 2 + (size_t)B * HC * 4;
    hipFuncSetAttribute((const void*)k_lstm_chunk,
                        hipFuncAttributeMaxDynamicSharedMemorySize, (int)ldsBytes);

    float* dstates = out + (size_t)B * S * H;  // h_n then c_n

    for (int layer = 0; layer < 2; ++layer) {
        const __bf16* xin  = layer ? xbf1 : xbf0;
        const __bf16* Wih  = layer ? Wt_ih1 : Wt_ih0;
        const __bf16* Whh  = layer ? Wt_hh1 : Wt_hh0;
        const float*  bias = layer ? bias1 : bias0;
        const int Kdim = layer ? H : I0;

        hipMemsetAsync(hbuf, 0, (size_t)2 * B * H * 2, stream);
        hipMemsetAsync(cst, 0, (size_t)B * H * 4, stream);

        for (int c = 0; c < S / CHUNK; ++c) {
            const int s0 = c * CHUNK;
            dim3 grid((CHUNK * B) / MT, G / NT);
            k_gemm_xg<<<grid, 256, 0, stream>>>(xin, Wih, bias, xg, Kdim, s0);
            hipMemsetAsync(counter, 0, 4, stream);
            k_lstm_chunk<<<NWG, 256, ldsBytes, stream>>>(
                xg, Whh, hbuf, cst,
                layer == 0 ? xbf1 : (__bf16*)nullptr,
                layer == 1 ? out : (float*)nullptr,
                dstates, counter, s0, layer);
        }
    }
}
